// GraphAwareEdgeModel_2370821947612
// MI455X (gfx1250) — matrix operand order repaired and checked by reading, not compiled, not run on hardware
//
#include <hip/hip_runtime.h>

#define NODE_F   128
#define EDGE_F   64
#define GRAPH_F  64
#define IN_F     384      // 2*NODE_F + EDGE_F + GRAPH_F
#define HID      256
#define MTILE    64
#define NEG_SLOPE 0.01f
#define LN_EPS   1e-5f

typedef __attribute__((ext_vector_type(16))) __bf16 bf16x16;
typedef __attribute__((ext_vector_type(8)))  __bf16 bf16x8;
typedef __attribute__((ext_vector_type(4)))  __bf16 bf16x4;
typedef __attribute__((ext_vector_type(8)))  float  f32x8;
typedef __attribute__((ext_vector_type(4)))  float  f32x4;

static __device__ __forceinline__ bf16x16 cat8(bf16x8 lo, bf16x8 hi) {
  return __builtin_shufflevector(lo, hi, 0,1,2,3,4,5,6,7,8,9,10,11,12,13,14,15);
}

// A fragment (16x32 bf16, row-major source, leading dim `ld` in elements).
// ISA 16-bit A layout: lanes 0-15 / 16-31 give K halves of 8; VGPR0-3 = K base,
// VGPR4-7 = K base + 16.
static __device__ __forceinline__ bf16x16 load_a_frag(const __bf16* base, int ld,
                                                      int row0, int k0, int lane) {
  const int idx = lane & 15, half = lane >> 4;
  const __bf16* p = base + (size_t)(row0 + idx) * ld + k0 + half * 8;
  bf16x8 lo = *(const bf16x8*)p;          // K = k0 + half*8 + 0..7
  bf16x8 hi = *(const bf16x8*)(p + 16);   // K = k0 + 16 + half*8 + 0..7
  return cat8(lo, hi);
}

// B fragment (32x16 KxN) from N-major (pre-transposed) weights Wt[N][K]:
// lanes 0-15 hold K=0..7,16..23, lanes 16-31 hold K=8..15,24..31 (matches A layout).
static __device__ __forceinline__ bf16x16 load_b_frag(const __bf16* baseT, int ld,
                                                      int n0, int k0, int lane) {
  const int idx = lane & 15, half = lane >> 4;
  const __bf16* p = baseT + (size_t)(n0 + idx) * ld + k0 + half * 8;
  bf16x8 lo = *(const bf16x8*)p;
  bf16x8 hi = *(const bf16x8*)(p + 16);
  return cat8(lo, hi);
}

// ---------------------------------------------------------------------------
// Prep: W1 [384][256] f32 -> W1t [256][384] bf16 ; W2 [256][256] -> W2t [256][256]
// ---------------------------------------------------------------------------
__global__ void prep_weights(const float* __restrict__ W1, const float* __restrict__ W2,
                             __bf16* __restrict__ W1t, __bf16* __restrict__ W2t) {
  const int i = blockIdx.x * 256 + threadIdx.x;
  if (i < IN_F * HID) {
    const int n = i / IN_F, k = i - n * IN_F;
    W1t[i] = (__bf16)W1[(size_t)k * HID + n];
  }
  const int j = i - IN_F * HID;
  if (j >= 0 && j < HID * HID) {
    const int n = j / HID, k = j - n * HID;
    W2t[j] = (__bf16)W2[(size_t)k * HID + n];
  }
}

// ---------------------------------------------------------------------------
// Fused: gather/concat -> GEMM1 -> bias -> LeakyReLU -> LayerNorm -> GEMM2 -> bias
// Block: 256 threads = 8 waves; 64 edges/block.
// Wave w: columns [w*32, w*32+32), all 4 M-subtiles -> 8 accumulators.
// Per K-tile: 4 A-frags (LDS) + 2 B-frags (global, L2-resident) -> 8 WMMAs.
// ---------------------------------------------------------------------------
__global__ __launch_bounds__(256) void edge_mlp_kernel(
    const float* __restrict__ src, const float* __restrict__ dst,
    const float* __restrict__ eattr, const float* __restrict__ u,
    const int* __restrict__ batch,
    const float* __restrict__ b1, const float* __restrict__ gamma,
    const float* __restrict__ beta, const float* __restrict__ b2,
    const __bf16* __restrict__ W1t, const __bf16* __restrict__ W2t,
    float* __restrict__ out, int E) {
  __shared__ __bf16 xs[MTILE][IN_F];   // 48 KB  staged input tile (bf16)
  __shared__ float  hf[MTILE][HID];    // 64 KB  GEMM1 result (f32) for LayerNorm
  __shared__ __bf16 hb[MTILE][HID];    // 32 KB  normalized h (bf16) for GEMM2
  __shared__ int    bidx[MTILE];

  const int tid  = threadIdx.x;
  const int lane = tid & 31;
  const int w    = tid >> 5;               // 8 waves
  const int e0   = blockIdx.x * MTILE;

  if (tid < MTILE) {
    const int e = e0 + tid;
    bidx[tid] = (e < E) ? batch[e] : 0;
  }
  __syncthreads();

  // ---- stage x = concat(src,dst,edge_attr,u[batch]) as bf16 -----------------
  for (int i4 = tid; i4 < MTILE * (IN_F / 4); i4 += 256) {
    const int row = i4 / (IN_F / 4);
    const int k   = (i4 - row * (IN_F / 4)) * 4;
    const int e   = e0 + row;
    f32x4 v = {0.f, 0.f, 0.f, 0.f};
    if (e < E) {
      if (k < NODE_F)
        v = *(const f32x4*)(src + (size_t)e * NODE_F + k);
      else if (k < 2 * NODE_F)
        v = *(const f32x4*)(dst + (size_t)e * NODE_F + (k - NODE_F));
      else if (k < 2 * NODE_F + EDGE_F)
        v = *(const f32x4*)(eattr + (size_t)e * EDGE_F + (k - 2 * NODE_F));
      else
        v = *(const f32x4*)(u + (size_t)bidx[row] * GRAPH_F + (k - 2 * NODE_F - EDGE_F));
    }
    bf16x4 bv = { (__bf16)v.x, (__bf16)v.y, (__bf16)v.z, (__bf16)v.w };
    *(bf16x4*)&xs[row][k] = bv;
  }
  __syncthreads();

  const int ncol0 = w * 32;  // wave's 32 columns of HID (2 N-tiles)

  // ---- GEMM1: h = x @ W1 ----------------------------------------------------
  f32x8 acc[8] = {};         // acc[ms*2+nt]: M-subtile ms (0..3), N-tile nt (0..1)
  for (int kt = 0; kt < IN_F; kt += 32) {
    bf16x16 a[4];
#pragma unroll
    for (int ms = 0; ms < 4; ++ms)
      a[ms] = load_a_frag(&xs[0][0], IN_F, ms * 16, kt, lane);
#pragma unroll
    for (int nt = 0; nt < 2; ++nt) {
      bf16x16 b = load_b_frag(W1t, IN_F, ncol0 + nt * 16, kt, lane);
#pragma unroll
      for (int ms = 0; ms < 4; ++ms)
        acc[ms * 2 + nt] = __builtin_amdgcn_wmma_f32_16x16x32_bf16(
            false, a[ms], false, b, (short)0, acc[ms * 2 + nt], false, false);
    }
  }

  // ---- epilogue1: +b1, LeakyReLU -> hf (f32) --------------------------------
  {
    const int idx = lane & 15, half = lane >> 4;
#pragma unroll
    for (int nt = 0; nt < 2; ++nt) {
      const int col = ncol0 + nt * 16 + idx;
      const float bias = b1[col];
#pragma unroll
      for (int ms = 0; ms < 4; ++ms) {
#pragma unroll
        for (int r = 0; r < 8; ++r) {
          float v = acc[ms * 2 + nt][r] + bias;
          v = (v >= 0.f) ? v : NEG_SLOPE * v;
          hf[ms * 16 + half * 8 + r][col] = v;
        }
      }
    }
  }
  __syncthreads();

  // ---- LayerNorm: wave w handles rows w*8 .. w*8+7 --------------------------
  {
    float gv[8], btv[8];
#pragma unroll
    for (int j = 0; j < 8; ++j) {
      gv[j]  = gamma[j * 32 + lane];
      btv[j] = beta[j * 32 + lane];
    }
#pragma unroll
    for (int rr = 0; rr < 8; ++rr) {
      const int row = w * 8 + rr;
      float s = 0.f, s2 = 0.f, vals[8];
#pragma unroll
      for (int j = 0; j < 8; ++j) {
        const float v = hf[row][j * 32 + lane];
        vals[j] = v; s += v; s2 += v * v;
      }
#pragma unroll
      for (int off = 16; off > 0; off >>= 1) {
        s  += __shfl_xor(s,  off, 32);
        s2 += __shfl_xor(s2, off, 32);
      }
      const float mu  = s * (1.f / HID);
      const float var = s2 * (1.f / HID) - mu * mu;
      const float rs  = rsqrtf(var + LN_EPS);
#pragma unroll
      for (int j = 0; j < 8; ++j) {
        const float v = (vals[j] - mu) * rs * gv[j] + btv[j];
        hb[row][j * 32 + lane] = (__bf16)v;
      }
    }
  }
  __syncthreads();

  // ---- GEMM2: out = hb @ W2 -------------------------------------------------
  f32x8 acc2[8] = {};
  for (int kt = 0; kt < HID; kt += 32) {
    bf16x16 a[4];
#pragma unroll
    for (int ms = 0; ms < 4; ++ms)
      a[ms] = load_a_frag(&hb[0][0], HID, ms * 16, kt, lane);
#pragma unroll
    for (int nt = 0; nt < 2; ++nt) {
      bf16x16 b = load_b_frag(W2t, HID, ncol0 + nt * 16, kt, lane);
#pragma unroll
      for (int ms = 0; ms < 4; ++ms)
        acc2[ms * 2 + nt] = __builtin_amdgcn_wmma_f32_16x16x32_bf16(
            false, a[ms], false, b, (short)0, acc2[ms * 2 + nt], false, false);
    }
  }

  // ---- epilogue2: +b2, store ------------------------------------------------
  {
    const int idx = lane & 15, half = lane >> 4;
#pragma unroll
    for (int nt = 0; nt < 2; ++nt) {
      const int col = ncol0 + nt * 16 + idx;
      const float bias = b2[col];
#pragma unroll
      for (int ms = 0; ms < 4; ++ms) {
#pragma unroll
        for (int r = 0; r < 8; ++r) {
          const int e = e0 + ms * 16 + half * 8 + r;
          if (e < E) out[(size_t)e * HID + col] = acc2[ms * 2 + nt][r] + bias;
        }
      }
    }
  }
}

// ---------------------------------------------------------------------------
extern "C" void kernel_launch(void* const* d_in, const int* in_sizes, int n_in,
                              void* d_out, int out_size, void* d_ws, size_t ws_size,
                              hipStream_t stream) {
  const float* src   = (const float*)d_in[0];
  const float* dst   = (const float*)d_in[1];
  const float* eattr = (const float*)d_in[2];
  const float* u     = (const float*)d_in[3];
  const int*   batch = (const int*)d_in[4];
  const float* W1    = (const float*)d_in[5];
  const float* b1    = (const float*)d_in[6];
  const float* gamma = (const float*)d_in[7];
  const float* beta  = (const float*)d_in[8];
  const float* W2    = (const float*)d_in[9];
  const float* b2    = (const float*)d_in[10];
  float* out = (float*)d_out;

  const int E = in_sizes[0] / NODE_F;

  __bf16* W1t = (__bf16*)d_ws;           // 256*384 bf16 = 192 KB
  __bf16* W2t = W1t + IN_F * HID;        // 256*256 bf16 = 128 KB

  const int prep_total = IN_F * HID + HID * HID;
  hipLaunchKernelGGL(prep_weights, dim3((prep_total + 255) / 256), dim3(256), 0, stream,
                     W1, W2, W1t, W2t);

  const int nblocks = (E + MTILE - 1) / MTILE;
  hipLaunchKernelGGL(edge_mlp_kernel, dim3(nblocks), dim3(256), 0, stream,
                     src, dst, eattr, u, batch, b1, gamma, beta, b2, W1t, W2t, out, E);
}
